// JointVQVAE_644245094987
// MI455X (gfx1250) — matrix operand order repaired and checked by reading, not compiled, not run on hardware
//
#include <hip/hip_runtime.h>
#include <hip/hip_bf16.h>

typedef __attribute__((ext_vector_type(16))) _Float16 v16h;
typedef __attribute__((ext_vector_type(8)))  _Float16 v8h;
typedef __attribute__((ext_vector_type(4)))  _Float16 v4h;
typedef __attribute__((ext_vector_type(8)))  float    v8f;

// ---------------------------------------------------------------------------
// WMMA fragment loaders (layouts per CDNA5 ISA 7.12.2, wave32)
// A 16x32 f16: lane (l16,hs) -> row=l16, K = {hs*8..hs*8+7, 16+hs*8..+7}
// B 32x16 f16 (stored transposed in LDS as [n][k]): col=l16, K = {hs*8..hs*8+7, 16+hs*8..+7}
// C/D 16x16 f32: row = v + 8*hs, col = l16
// LDS row strides are padded (+8 halfs) so the 16 lanes of a fragment load
// hit distinct banks (stride 144B -> bank step 36, conflict-free).
// ---------------------------------------------------------------------------
__device__ __forceinline__ v16h frag_a(const _Float16* p, int hs) {
  v8h lo = *(const v8h*)(p + hs * 8);
  v8h hi = *(const v8h*)(p + 16 + hs * 8);
  v16h a;
#pragma unroll
  for (int i = 0; i < 8; ++i) { a[i] = lo[i]; a[i + 8] = hi[i]; }
  return a;
}
__device__ __forceinline__ v16h frag_b(const _Float16* p, int hs) {
  v8h lo = *(const v8h*)(p + hs * 8);
  v8h hi = *(const v8h*)(p + 16 + hs * 8);
  v16h b;
#pragma unroll
  for (int i = 0; i < 8; ++i) { b[i] = lo[i]; b[i + 8] = hi[i]; }
  return b;
}
__device__ __forceinline__ float gelu_f(float x) {
  const float c = 0.7978845608028654f;  // sqrt(2/pi), jax approximate gelu
  return 0.5f * x * (1.0f + tanhf(c * (x + 0.044715f * x * x * x)));
}
__device__ __forceinline__ float block_reduce_sum(float v, float* red) {
  int t = threadIdx.x;
  red[t] = v;
  __syncthreads();
#pragma unroll
  for (int s = 128; s > 0; s >>= 1) {
    if (t < s) red[t] += red[t + s];
    __syncthreads();
  }
  float r = red[0];
  __syncthreads();
  return r;
}

// ---------------------------------------------------------------------------
// Register-staged tile movers (16B global loads; f32->f16 convert on the fly)
// ---------------------------------------------------------------------------
template <int BM, int BK, int NV>
__device__ __forceinline__ void load_A_f16(v8h (&r)[NV], const _Float16* __restrict__ A,
                                           int m0, int k0, int K, int t) {
  constexpr int VPR = BK / 8;
#pragma unroll
  for (int i = 0; i < NV; ++i) {
    int q = t + i * 256;
    int rr = q / VPR, c8 = (q % VPR) * 8;
    r[i] = *(const v8h*)&A[(size_t)(m0 + rr) * K + k0 + c8];
  }
}
template <int BM, int BK, int SK, int NV>
__device__ __forceinline__ void store_A(const v8h (&r)[NV], _Float16* __restrict__ dst, int t) {
  constexpr int VPR = BK / 8;
#pragma unroll
  for (int i = 0; i < NV; ++i) {
    int q = t + i * 256;
    int rr = q / VPR, c8 = (q % VPR) * 8;
    *(v8h*)&dst[rr * SK + c8] = r[i];
  }
}
template <int BN, int BK, int NQ>
__device__ __forceinline__ void load_B(float4 (&r)[NQ], const float* __restrict__ B,
                                       int n0, int k0, int N, int t) {
  constexpr int QPR = BN / 4;
#pragma unroll
  for (int i = 0; i < NQ; ++i) {
    int q = t + i * 256;
    int k = q / QPR, n4 = (q % QPR) * 4;
    r[i] = *(const float4*)&B[(size_t)(k0 + k) * N + n0 + n4];
  }
}
template <int BN, int BK, int SK, int NQ>
__device__ __forceinline__ void store_B(const float4 (&r)[NQ], _Float16* __restrict__ dst, int t) {
  constexpr int QPR = BN / 4;
#pragma unroll
  for (int i = 0; i < NQ; ++i) {
    int q = t + i * 256;
    int k = q / QPR, n4 = (q % QPR) * 4;
    dst[(n4 + 0) * SK + k] = (_Float16)r[i].x;
    dst[(n4 + 1) * SK + k] = (_Float16)r[i].y;
    dst[(n4 + 2) * SK + k] = (_Float16)r[i].z;
    dst[(n4 + 3) * SK + k] = (_Float16)r[i].w;
  }
}
template <int BM, int BK, int SK>
__device__ __forceinline__ void stage_A_f32(_Float16* __restrict__ dst,
                                            const float* __restrict__ A,
                                            int m0, int k0, int K, int t) {
  constexpr int QPR = BK / 4;
#pragma unroll
  for (int q = t; q < BM * QPR; q += 256) {
    int r = q / QPR, c4 = (q % QPR) * 4;
    float4 f = *(const float4*)&A[(size_t)(m0 + r) * K + k0 + c4];
    v4h h;
    h[0] = (_Float16)f.x; h[1] = (_Float16)f.y;
    h[2] = (_Float16)f.z; h[3] = (_Float16)f.w;
    *(v4h*)&dst[r * SK + c4] = h;
  }
}

// ---------------------------------------------------------------------------
// Generic WMMA GEMM: Out[M,N] = act( A[M,K] @ B[K,N] + bias )
// 256 threads = 8 waves; double-buffered LDS; next tile is prefetched into
// registers BEFORE the WMMA block and drained to LDS after it, so global
// latency is hidden under matrix math. One barrier per BK stage.
// ---------------------------------------------------------------------------
template <int BM, int BN, int BK, int WAVES_M, int WAVES_N,
          bool A_F32, bool DO_GELU, bool OUT_F16>
__global__ __launch_bounds__(256) void gemm_kernel(
    const void* __restrict__ Ap, const float* __restrict__ B,
    const float* __restrict__ bias, void* __restrict__ Outp,
    int M, int N, int K) {
  constexpr int WM = BM / 16 / WAVES_M;
  constexpr int WN = BN / 16 / WAVES_N;
  constexpr int SK = BK + 8;                    // padded LDS stride (halfs)
  constexpr int NVA = BM * BK / (8 * 256);      // v8h per thread for A tile
  constexpr int NQB = BK * BN / (4 * 256);      // float4 per thread for B tile
  __shared__ __align__(32) _Float16 sA[2][BM * SK];
  __shared__ __align__(32) _Float16 sB[2][BN * SK];

  int t = threadIdx.x;
  int lane = t & 31, wid = t >> 5;
  int hs = lane >> 4, l16 = lane & 15;
  int wave_m = wid % WAVES_M, wave_n = wid / WAVES_M;
  int m0 = blockIdx.y * BM, n0 = blockIdx.x * BN;

  v8f acc[WM][WN];
  v8f zacc = {};
#pragma unroll
  for (int i = 0; i < WM; ++i)
#pragma unroll
    for (int j = 0; j < WN; ++j) acc[i][j] = zacc;

  // stage 0
  if (A_F32) {
    stage_A_f32<BM, BK, SK>(sA[0], (const float*)Ap, m0, 0, K, t);
  } else {
    v8h ra0[NVA];
    load_A_f16<BM, BK, NVA>(ra0, (const _Float16*)Ap, m0, 0, K, t);
    store_A<BM, BK, SK, NVA>(ra0, sA[0], t);
  }
  {
    float4 rb0[NQB];
    load_B<BN, BK, NQB>(rb0, B, n0, 0, N, t);
    store_B<BN, BK, SK, NQB>(rb0, sB[0], t);
  }
  __syncthreads();

  const int nk = K / BK;
  for (int kt = 0; kt < nk; ++kt) {
    int cur = kt & 1;
    bool pre = (kt + 1 < nk);
    v8h ra[NVA];
    float4 rb[NQB];
    if (pre) {  // issue next tile's global loads before the WMMA block
      if (!A_F32) load_A_f16<BM, BK, NVA>(ra, (const _Float16*)Ap, m0, (kt + 1) * BK, K, t);
      load_B<BN, BK, NQB>(rb, B, n0, (kt + 1) * BK, N, t);
    }
    const _Float16* cA = sA[cur];
    const _Float16* cB = sB[cur];
#pragma unroll
    for (int kk = 0; kk < BK; kk += 32) {
      v16h afr[WM];
#pragma unroll
      for (int im = 0; im < WM; ++im)
        afr[im] = frag_a(&cA[(wave_m * WM * 16 + im * 16 + l16) * SK + kk], hs);
#pragma unroll
      for (int in = 0; in < WN; ++in) {
        v16h bfr = frag_b(&cB[(wave_n * WN * 16 + in * 16 + l16) * SK + kk], hs);
#pragma unroll
        for (int im = 0; im < WM; ++im)
          acc[im][in] = __builtin_amdgcn_wmma_f32_16x16x32_f16(
              false, afr[im], false, bfr, (short)0, acc[im][in], false, false);
      }
    }
    if (pre) {  // drain registers into the other LDS buffer
      if (A_F32) stage_A_f32<BM, BK, SK>(sA[cur ^ 1], (const float*)Ap, m0, (kt + 1) * BK, K, t);
      else       store_A<BM, BK, SK, NVA>(ra, sA[cur ^ 1], t);
      store_B<BN, BK, SK, NQB>(rb, sB[cur ^ 1], t);
    }
    __syncthreads();
  }

#pragma unroll
  for (int im = 0; im < WM; ++im) {
#pragma unroll
    for (int in = 0; in < WN; ++in) {
      int gc = n0 + wave_n * WN * 16 + in * 16 + l16;
      float bv = bias[gc];
#pragma unroll
      for (int v = 0; v < 8; ++v) {
        int gr = m0 + wave_m * WM * 16 + im * 16 + v + 8 * hs;
        float x = acc[im][in][v] + bv;
        if (DO_GELU) x = gelu_f(x);
        if (OUT_F16) ((_Float16*)Outp)[(size_t)gr * N + gc] = (_Float16)x;
        else         ((float*)Outp)[(size_t)gr * N + gc] = x;
      }
    }
  }
}

// ---------------------------------------------------------------------------
// Row L2-normalize (256-wide rows)
// ---------------------------------------------------------------------------
__global__ __launch_bounds__(256) void norm_f32_to_f16_kernel(
    const float* __restrict__ in, _Float16* __restrict__ out) {
  __shared__ float red[256];
  size_t row = blockIdx.x;
  int t = threadIdx.x;
  float v = in[row * 256 + t];
  float s = block_reduce_sum(v * v, red);
  float inv = 1.0f / (sqrtf(s) + 1e-8f);
  out[row * 256 + t] = (_Float16)(v * inv);
}

__global__ __launch_bounds__(256) void norm_f16_inplace_kernel(_Float16* __restrict__ x) {
  __shared__ float red[256];
  size_t row = blockIdx.x;
  int t = threadIdx.x;
  float v = (float)x[row * 256 + t];
  float s = block_reduce_sum(v * v, red);
  float inv = 1.0f / (sqrtf(s) + 1e-8f);
  x[row * 256 + t] = (_Float16)(v * inv);
}

// ---------------------------------------------------------------------------
// Code search: scores = z_e[64,256] @ cbn^T, fused running argmax over 1024
// codes (unit vectors -> argmin dist == argmax dot). 8 waves: 4 (rows) x 2.
// Next codebook chunk is prefetched into registers during the current
// chunk's WMMAs and drained after the post-compute barrier.
// ---------------------------------------------------------------------------
__global__ __launch_bounds__(256) void dist_argmax_kernel(
    const _Float16* __restrict__ ze, const _Float16* __restrict__ cbn,
    int* __restrict__ idx_out, float* __restrict__ idxf_out) {
  constexpr int SK = 264;                          // padded stride
  __shared__ __align__(32) _Float16 sA[64 * SK];   // ~33 KB, staged once
  __shared__ __align__(32) _Float16 sB[32 * SK];   // ~17 KB code chunk
  __shared__ float sS[64 * 32];                    // 8 KB scores
  int t = threadIdx.x;
  int m0 = blockIdx.x * 64;
  int lane = t & 31, wid = t >> 5, hs = lane >> 4, l16 = lane & 15;
  int wave_m = wid & 3, wave_n = wid >> 2;

#pragma unroll
  for (int q = t; q < 64 * 32; q += 256) {  // 32 v8h per 256-wide row
    int r = q >> 5, c8 = (q & 31) * 8;
    *(v8h*)&sA[r * SK + c8] = *(const v8h*)&ze[(size_t)(m0 + r) * 256 + c8];
  }
#pragma unroll
  for (int q = t; q < 32 * 32; q += 256) {  // chunk 0
    int r = q >> 5, c8 = (q & 31) * 8;
    *(v8h*)&sB[r * SK + c8] = *(const v8h*)&cbn[(size_t)r * 256 + c8];
  }
  float best = -1e30f;
  int bidx = 0;
  __syncthreads();

  for (int c0 = 0; c0 < 1024; c0 += 32) {
    bool pre = (c0 + 32 < 1024);
    v8h rb[4];
    if (pre) {  // prefetch next chunk while computing this one
#pragma unroll
      for (int i = 0; i < 4; ++i) {
        int q = t + i * 256;
        int r = q >> 5, c8 = (q & 31) * 8;
        rb[i] = *(const v8h*)&cbn[(size_t)(c0 + 32 + r) * 256 + c8];
      }
    }
    v8f acc = {};
#pragma unroll
    for (int k0 = 0; k0 < 256; k0 += 32) {
      v16h a = frag_a(&sA[(wave_m * 16 + l16) * SK + k0], hs);
      v16h b = frag_b(&sB[(wave_n * 16 + l16) * SK + k0], hs);
      acc = __builtin_amdgcn_wmma_f32_16x16x32_f16(false, a, false, b, (short)0, acc, false, false);
    }
#pragma unroll
    for (int v = 0; v < 8; ++v)
      sS[(wave_m * 16 + v + 8 * hs) * 32 + wave_n * 16 + l16] = acc[v];
    __syncthreads();  // all WMMA reads of sB done; sS complete
    if (t < 64) {
      for (int c = 0; c < 32; ++c) {
        float s = sS[t * 32 + c];
        if (s > best) { best = s; bidx = c0 + c; }  // strict > keeps first tie
      }
    }
    if (pre) {
#pragma unroll
      for (int i = 0; i < 4; ++i) {
        int q = t + i * 256;
        int r = q >> 5, c8 = (q & 31) * 8;
        *(v8h*)&sB[r * SK + c8] = rb[i];
      }
    }
    __syncthreads();
  }
  if (t < 64) {
    int row = m0 + t;
    idx_out[row] = bidx;
    idxf_out[row] = (float)bidx;
  }
}

// Gather z_q = cbn[idx] and accumulate sum((z_e - z_q)^2)
__global__ __launch_bounds__(256) void gather_loss_kernel(
    const int* __restrict__ idx, const _Float16* __restrict__ cbn,
    const _Float16* __restrict__ ze, _Float16* __restrict__ zq,
    float* __restrict__ acc) {
  __shared__ float red[256];
  size_t row = blockIdx.x;
  int t = threadIdx.x;
  int ci = idx[row];
  _Float16 q = cbn[(size_t)ci * 256 + t];
  float e = (float)ze[row * 256 + t];
  zq[row * 256 + t] = q;
  float d = e - (float)q;
  float s = block_reduce_sum(d * d, red);
  if (t == 0) atomicAdd(acc + 1, s);
}

__global__ __launch_bounds__(256) void recon_loss_kernel(
    const float* __restrict__ action, const float* __restrict__ ahat,
    const float* __restrict__ w, float* __restrict__ acc) {
  __shared__ float red[256];
  size_t i = (size_t)blockIdx.x * 256 + threadIdx.x;
  const size_t total = (size_t)65536 * 32;
  float v = 0.f;
  for (size_t j = i; j < total; j += (size_t)gridDim.x * 256) {
    float d = action[j] - ahat[j];
    v += d * d * w[j & 31];
  }
  float s = block_reduce_sum(v, red);
  if (threadIdx.x == 0) atomicAdd(acc + 0, s);
}

__global__ void init_kernel(float* acc) {
  if (threadIdx.x < 2) acc[threadIdx.x] = 0.f;
}

__global__ void finalize_kernel(const float* __restrict__ acc, float* __restrict__ out) {
  float recon = acc[0] / (65536.0f * 32.0f);
  float vq    = acc[1] / (65536.0f * 256.0f);
  out[0] = recon;
  out[1] = vq;   // codebook_loss (forward value)
  out[2] = vq;   // commit_loss == codebook_loss numerically in forward
}

// ---------------------------------------------------------------------------
extern "C" void kernel_launch(void* const* d_in, const int* in_sizes, int n_in,
                              void* d_out, int out_size, void* d_ws, size_t ws_size,
                              hipStream_t stream) {
  const int N = 65536, A = 32;
  const float* action = (const float*)d_in[0];
  const float* dimw   = (const float*)d_in[1];
  const float* ew0 = (const float*)d_in[2];  const float* eb0 = (const float*)d_in[3];
  const float* ew1 = (const float*)d_in[4];  const float* eb1 = (const float*)d_in[5];
  const float* ew2 = (const float*)d_in[6];  const float* eb2 = (const float*)d_in[7];
  const float* dw0 = (const float*)d_in[8];  const float* db0 = (const float*)d_in[9];
  const float* dw1 = (const float*)d_in[10]; const float* db1 = (const float*)d_in[11];
  const float* dw2 = (const float*)d_in[12]; const float* db2 = (const float*)d_in[13];
  const float* cb  = (const float*)d_in[14];

  char* ws = (char*)d_ws;
  float*    acc    = (float*)(ws);                                     // 2 floats
  int*      idxbuf = (int*)(ws + 1024);                                // N ints
  _Float16* cbn    = (_Float16*)(ws + 1024 + (size_t)N * 4);           // 1024*256
  _Float16* h1     = (_Float16*)((char*)cbn + (size_t)1024 * 256 * 2); // N*512
  _Float16* h2     = (_Float16*)((char*)h1 + (size_t)N * 512 * 2);     // N*1024
  _Float16* ze     = (_Float16*)((char*)h2 + (size_t)N * 1024 * 2);    // N*256
  _Float16* zq     = (_Float16*)((char*)ze + (size_t)N * 256 * 2);     // N*256

  float* out_ahat = (float*)d_out;
  float* out_idx  = out_ahat + (size_t)N * A;
  float* out_scal = out_idx + N;

  init_kernel<<<1, 32, 0, stream>>>(acc);
  norm_f32_to_f16_kernel<<<1024, 256, 0, stream>>>(cb, cbn);

  // encoder: 32 -> 512 -> 1024 -> 256
  gemm_kernel<128, 64, 32, 4, 2, true,  true,  true ><<<dim3(8,  N / 128), 256, 0, stream>>>(action, ew0, eb0, h1, N, 512, 32);
  gemm_kernel<128, 64, 64, 4, 2, false, true,  true ><<<dim3(16, N / 128), 256, 0, stream>>>(h1, ew1, eb1, h2, N, 1024, 512);
  gemm_kernel<128, 64, 64, 4, 2, false, false, true ><<<dim3(4,  N / 128), 256, 0, stream>>>(h2, ew2, eb2, ze, N, 256, 1024);
  norm_f16_inplace_kernel<<<N, 256, 0, stream>>>(ze);

  // code search + gather + vq loss
  dist_argmax_kernel<<<N / 64, 256, 0, stream>>>(ze, cbn, idxbuf, out_idx);
  gather_loss_kernel<<<N, 256, 0, stream>>>(idxbuf, cbn, ze, zq, acc);

  // decoder: 256 -> 1024 -> 512 -> 32   (reuse h2, h1 buffers)
  gemm_kernel<128, 64, 64, 4, 2, false, true,  true ><<<dim3(16, N / 128), 256, 0, stream>>>(zq, dw0, db0, h2, N, 1024, 256);
  gemm_kernel<128, 64, 64, 4, 2, false, true,  true ><<<dim3(8,  N / 128), 256, 0, stream>>>(h2, dw1, db1, h1, N, 512, 1024);
  gemm_kernel<128, 32, 64, 8, 1, false, false, false><<<dim3(1,  N / 128), 256, 0, stream>>>(h1, dw2, db2, out_ahat, N, 32, 512);

  recon_loss_kernel<<<1024, 256, 0, stream>>>(action, out_ahat, dimw, acc);
  finalize_kernel<<<1, 1, 0, stream>>>(acc, out_scal);
}
